// TorchL2ROneLayerEncDecGruSeqPredResInfer_75179107549684
// MI455X (gfx1250) — matrix operand order repaired and checked by reading, not compiled, not run on hardware
//
#include <hip/hip_runtime.h>
#include <hip/hip_bf16.h>
#include <math.h>

// ---------------- model dims ----------------
#define BB    1024
#define TT    64
#define EMB   256
#define HID   512
#define G3    (3 * HID)      // 1536
#define NCLS  128
#define CLSL  512
#define STEPS 63             // MAX_STEPS - 1

typedef __bf16 bf16_t;
typedef __attribute__((ext_vector_type(16))) __bf16 v16bf;
typedef __attribute__((ext_vector_type(8)))  float  v8f;

// ---------------- WMMA helpers ----------------
// A-matrix 16x32 bf16 fragment (ISA 7.12.2): lanes 0-15 hold K {0..7,16..23},
// lanes 16-31 hold K {8..15,24..31}; packed 2 bf16 per VGPR.
__device__ __forceinline__ v16bf load_frag_a(const bf16_t* rp, int lane) {
  union { v16bf v; unsigned u[8]; } f;
  const bf16_t* p = rp + ((lane & 16) ? 8 : 0);
#pragma unroll
  for (int j = 0; j < 4; ++j) {
    f.u[j]     = *(const unsigned*)(p + 2 * j);
    f.u[j + 4] = *(const unsigned*)(p + 16 + 2 * j);
  }
  return f.v;
}

// B-matrix 32x16 bf16 fragment: lane = output column; lanes 0-15 hold K {0..7,16..23},
// lanes 16-31 hold K {8..15,24..31} (same K order as the A fragment).
__device__ __forceinline__ v16bf load_frag_b(const bf16_t* rp, int lane) {
  union { v16bf v; unsigned u[8]; } f;
  const bf16_t* p = rp + ((lane & 16) ? 8 : 0);
#pragma unroll
  for (int j = 0; j < 4; ++j) {
    f.u[j]     = *(const unsigned*)(p + 2 * j);
    f.u[j + 4] = *(const unsigned*)(p + 16 + 2 * j);
  }
  return f.v;
}

__device__ __forceinline__ v8f wmma_bf16(v16bf a, v16bf b, v8f c) {
  return __builtin_amdgcn_wmma_f32_16x16x32_bf16(
      /*neg_a=*/false, a, /*neg_b=*/false, b,
      /*c_mod=*/(short)0, c, /*reuse_a=*/false, /*reuse_b=*/false);
}

__device__ __forceinline__ float sigm(float x) { return 1.0f / (1.0f + __expf(-x)); }
__device__ __forceinline__ float gelu_exact(float x) {
  return 0.5f * x * (1.0f + erff(x * 0.70710678118654752f));
}

// ---------------- utility kernels ----------------
__global__ void k_f2bf(const float* __restrict__ src, bf16_t* __restrict__ dst, int n) {
  int i = blockIdx.x * blockDim.x + threadIdx.x;
  if (i < n) dst[i] = (bf16_t)src[i];
}

__global__ void k_init_state(float* __restrict__ hf, bf16_t* __restrict__ hbf,
                             int n, int* __restrict__ x0, int nx) {
  int i = blockIdx.x * blockDim.x + threadIdx.x;
  if (i < n) { hf[i] = 0.0f; hbf[i] = (bf16_t)0.0f; }
  if (i < nx) x0[i] = 0;
}

// ---------------- fused GRU step ----------------
// One wave computes a 16x16 tile of h_out: accumulates r,z (gi+gh), i_n (gi),
// h_n (gh) via WMMA, then gate math + h update. Used for encoder and decoder.
// grid: 256 blocks x 256 threads = 2048 waves = 64 (M tiles) * 32 (J tiles).
__global__ __launch_bounds__(256)
void k_gru_step(const int* __restrict__ tokens, int tok_stride,
                const bf16_t* __restrict__ emb,        // [vocab, 256]
                const bf16_t* __restrict__ Wih,        // [1536, 256]
                const bf16_t* __restrict__ Whh,        // [1536, 512]
                const float* __restrict__ bih,         // [1536]
                const float* __restrict__ bhh,         // [1536]
                const float* __restrict__ h_in_f,      // [B, 512]
                const bf16_t* __restrict__ h_in_bf,    // [B, 512]
                float* __restrict__ h_out_f,
                bf16_t* __restrict__ h_out_bf,
                const int* __restrict__ lens, int t,   // take-last (encoder only)
                float* __restrict__ last_f, bf16_t* __restrict__ last_bf)
{
  const int lane = threadIdx.x & 31;
  const int wave = (blockIdx.x * blockDim.x + threadIdx.x) >> 5;
  const int mt = wave >> 5;          // 0..63
  const int jt = wave & 31;          // 0..31
  const int row_a = mt * 16 + (lane & 15);     // batch row this lane serves (A)
  const int col_b = jt * 16 + (lane & 15);     // output column / weight row (B)

  v8f acc_r  = {0.f, 0.f, 0.f, 0.f, 0.f, 0.f, 0.f, 0.f};
  v8f acc_z  = acc_r;
  v8f acc_in = acc_r;   // gi n-part
  v8f acc_hn = acc_r;   // gh n-part

  // ---- gi = xemb @ Wih^T (token gather folded into A load) ----
  const int tok = tokens[(size_t)row_a * tok_stride];
  const bf16_t* arow = emb + (size_t)tok * EMB;
  const bf16_t* wr = Wih + (size_t)col_b * EMB;
  const bf16_t* wz = Wih + (size_t)(col_b + HID) * EMB;
  const bf16_t* wn = Wih + (size_t)(col_b + 2 * HID) * EMB;
#pragma unroll
  for (int k0 = 0; k0 < EMB; k0 += 32) {
    v16bf a = load_frag_a(arow + k0, lane);
    acc_r  = wmma_bf16(a, load_frag_b(wr + k0, lane), acc_r);
    acc_z  = wmma_bf16(a, load_frag_b(wz + k0, lane), acc_z);
    acc_in = wmma_bf16(a, load_frag_b(wn + k0, lane), acc_in);
  }

  // ---- gh = h @ Whh^T ----
  const bf16_t* hrow = h_in_bf + (size_t)row_a * HID;
  const bf16_t* vr = Whh + (size_t)col_b * HID;
  const bf16_t* vz = Whh + (size_t)(col_b + HID) * HID;
  const bf16_t* vn = Whh + (size_t)(col_b + 2 * HID) * HID;
#pragma unroll 4
  for (int k0 = 0; k0 < HID; k0 += 32) {
    v16bf a = load_frag_a(hrow + k0, lane);
    acc_r  = wmma_bf16(a, load_frag_b(vr + k0, lane), acc_r);
    acc_z  = wmma_bf16(a, load_frag_b(vz + k0, lane), acc_z);
    acc_hn = wmma_bf16(a, load_frag_b(vn + k0, lane), acc_hn);
  }

  // ---- gate math + write ----
  const int j_g = jt * 16 + (lane & 15);
  const float br  = bih[j_g] + bhh[j_g];
  const float bz  = bih[HID + j_g] + bhh[HID + j_g];
  const float bin = bih[2 * HID + j_g];
  const float bhn = bhh[2 * HID + j_g];
  const int m_hi = (lane & 16) ? 8 : 0;
#pragma unroll
  for (int p = 0; p < 8; ++p) {
    const int m_g = mt * 16 + p + m_hi;
    const float r  = sigm(acc_r[p] + br);
    const float z  = sigm(acc_z[p] + bz);
    const float nn = tanhf(acc_in[p] + bin + r * (acc_hn[p] + bhn));
    const float hp = h_in_f[(size_t)m_g * HID + j_g];
    const float h  = (1.0f - z) * nn + z * hp;
    h_out_f[(size_t)m_g * HID + j_g]  = h;
    h_out_bf[(size_t)m_g * HID + j_g] = (bf16_t)h;
    if (lens != nullptr) {
      if (lens[m_g] - 1 == t) {
        last_f[(size_t)m_g * HID + j_g]  = h;
        last_bf[(size_t)m_g * HID + j_g] = (bf16_t)h;
      }
    }
  }
}

// ---------------- fused classifier + argmax ----------------
// One block per 16 batch rows. Phase 1: hidden = gelu(h @ W1^T + b1) -> LDS.
// Phase 2: logits = hidden @ W2^T + b2 -> d_out + LDS. Phase 3: argmax.
#define LDH 520  // padded bf16 row stride for LDS hidden
__global__ __launch_bounds__(256)
void k_classifier(const bf16_t* __restrict__ hbf,     // [B, 512]
                  const bf16_t* __restrict__ W1,      // [512, 512]
                  const float* __restrict__ b1,       // [512]
                  const bf16_t* __restrict__ W2,      // [128, 512]
                  const float* __restrict__ b2,       // [128]
                  float* __restrict__ out, int step,
                  int* __restrict__ x_next)
{
  __shared__ __align__(16) bf16_t s_hid[16 * LDH];
  __shared__ __align__(16) float  s_lg[16 * NCLS];

  const int lane = threadIdx.x & 31;
  const int w    = threadIdx.x >> 5;   // 0..7
  const int m0   = blockIdx.x * 16;
  const int m_hi = (lane & 16) ? 8 : 0;
  const int nl   = lane & 15;

  // ---- phase 1: hidden (each wave: 4 N-tiles = 64 columns) ----
  {
    v8f acc[4];
#pragma unroll
    for (int q = 0; q < 4; ++q) acc[q] = (v8f){0.f,0.f,0.f,0.f,0.f,0.f,0.f,0.f};
    const bf16_t* arow = hbf + (size_t)(m0 + nl) * HID;
#pragma unroll 4
    for (int k0 = 0; k0 < HID; k0 += 32) {
      v16bf a = load_frag_a(arow + k0, lane);
#pragma unroll
      for (int q = 0; q < 4; ++q) {
        const int col = (w * 4 + q) * 16 + nl;       // W1 row
        acc[q] = wmma_bf16(a, load_frag_b(W1 + (size_t)col * HID + k0, lane), acc[q]);
      }
    }
#pragma unroll
    for (int q = 0; q < 4; ++q) {
      const int c = (w * 4 + q) * 16 + nl;
      const float bc = b1[c];
#pragma unroll
      for (int p = 0; p < 8; ++p) {
        const int m = p + m_hi;
        s_hid[m * LDH + c] = (bf16_t)gelu_exact(acc[q][p] + bc);
      }
    }
  }
  __syncthreads();

  // ---- phase 2: logits (each wave: one 16-col tile of 128) ----
  {
    v8f acc = (v8f){0.f,0.f,0.f,0.f,0.f,0.f,0.f,0.f};
    const int c = w * 16 + nl;                        // W2 row / logit column
    const bf16_t* arow = s_hid + nl * LDH;            // A from LDS
#pragma unroll 4
    for (int k0 = 0; k0 < HID; k0 += 32) {
      v16bf a = load_frag_a(arow + k0, lane);
      acc = wmma_bf16(a, load_frag_b(W2 + (size_t)c * HID + k0, lane), acc);
    }
    const float bc = b2[c];
#pragma unroll
    for (int p = 0; p < 8; ++p) {
      const int m = p + m_hi;
      const float v = acc[p] + bc;
      s_lg[m * NCLS + c] = v;
      out[((size_t)(m0 + m) * STEPS + step) * NCLS + c] = v;
    }
  }
  __syncthreads();

  // ---- phase 3: first-occurrence argmax per row ----
  if (threadIdx.x < 16) {
    const int row = threadIdx.x;
    float best = s_lg[row * NCLS];
    int bi = 0;
    for (int c = 1; c < NCLS; ++c) {
      const float v = s_lg[row * NCLS + c];
      if (v > best) { best = v; bi = c; }
    }
    x_next[m0 + row] = bi;
  }
}

// ---------------- host launcher ----------------
extern "C" void kernel_launch(void* const* d_in, const int* in_sizes, int n_in,
                              void* d_out, int out_size, void* d_ws, size_t ws_size,
                              hipStream_t stream) {
  const int*   input_ids  = (const int*)d_in[0];
  const int*   input_lens = (const int*)d_in[1];
  const float* enc_emb_W  = (const float*)d_in[2];
  const float* dec_emb_W  = (const float*)d_in[3];
  const float* enc_Wih    = (const float*)d_in[4];
  const float* enc_Whh    = (const float*)d_in[5];
  const float* enc_bih    = (const float*)d_in[6];
  const float* enc_bhh    = (const float*)d_in[7];
  const float* dec_Wih    = (const float*)d_in[8];
  const float* dec_Whh    = (const float*)d_in[9];
  const float* dec_bih    = (const float*)d_in[10];
  const float* dec_bhh    = (const float*)d_in[11];
  const float* cls_W1     = (const float*)d_in[12];
  const float* cls_b1     = (const float*)d_in[13];
  const float* cls_W2     = (const float*)d_in[14];
  const float* cls_b2     = (const float*)d_in[15];
  float* out = (float*)d_out;

  // workspace bump allocator (256B aligned)
  char* base = (char*)d_ws;
  size_t off = 0;
  auto alloc = [&](size_t bytes) -> void* {
    void* p = base + off;
    off = (off + bytes + 255) & ~(size_t)255;
    return p;
  };
  bf16_t* encEmb_bf = (bf16_t*)alloc(128 * EMB * sizeof(bf16_t));
  bf16_t* decEmb_bf = (bf16_t*)alloc(128 * EMB * sizeof(bf16_t));
  bf16_t* encWih_bf = (bf16_t*)alloc((size_t)G3 * EMB * sizeof(bf16_t));
  bf16_t* encWhh_bf = (bf16_t*)alloc((size_t)G3 * HID * sizeof(bf16_t));
  bf16_t* decWih_bf = (bf16_t*)alloc((size_t)G3 * EMB * sizeof(bf16_t));
  bf16_t* decWhh_bf = (bf16_t*)alloc((size_t)G3 * HID * sizeof(bf16_t));
  bf16_t* W1_bf     = (bf16_t*)alloc((size_t)CLSL * HID * sizeof(bf16_t));
  bf16_t* W2_bf     = (bf16_t*)alloc((size_t)NCLS * HID * sizeof(bf16_t));
  float*  HF[2];  bf16_t* HB[2];
  float*  DF[2];  bf16_t* DB[2];
  for (int i = 0; i < 2; ++i) {
    HF[i] = (float*)alloc((size_t)BB * HID * sizeof(float));
    HB[i] = (bf16_t*)alloc((size_t)BB * HID * sizeof(bf16_t));
  }
  for (int i = 0; i < 2; ++i) {
    DF[i] = (float*)alloc((size_t)BB * HID * sizeof(float));
    DB[i] = (bf16_t*)alloc((size_t)BB * HID * sizeof(bf16_t));
  }
  int* XB[2];
  XB[0] = (int*)alloc(BB * sizeof(int));
  XB[1] = (int*)alloc(BB * sizeof(int));
  (void)ws_size; (void)in_sizes; (void)n_in; (void)out_size;

  auto cvt = [&](const float* s, bf16_t* d, int n) {
    k_f2bf<<<(n + 255) / 256, 256, 0, stream>>>(s, d, n);
  };
  cvt(enc_emb_W, encEmb_bf, 128 * EMB);
  cvt(dec_emb_W, decEmb_bf, 128 * EMB);
  cvt(enc_Wih,   encWih_bf, G3 * EMB);
  cvt(enc_Whh,   encWhh_bf, G3 * HID);
  cvt(dec_Wih,   decWih_bf, G3 * EMB);
  cvt(dec_Whh,   decWhh_bf, G3 * HID);
  cvt(cls_W1,    W1_bf,     CLSL * HID);
  cvt(cls_W2,    W2_bf,     NCLS * HID);

  // h0 = 0, x0 = 0
  {
    const int n = BB * HID;
    k_init_state<<<(n + 255) / 256, 256, 0, stream>>>(HF[0], HB[0], n, XB[0], BB);
  }

  // ---- encoder scan (take-last scattered into decoder init buffers) ----
  for (int t = 0; t < TT; ++t) {
    const int in = t & 1, outb = in ^ 1;
    k_gru_step<<<256, 256, 0, stream>>>(
        input_ids + t, TT, encEmb_bf, encWih_bf, encWhh_bf, enc_bih, enc_bhh,
        HF[in], HB[in], HF[outb], HB[outb],
        input_lens, t, DF[0], DB[0]);
  }

  // ---- autoregressive decoder ----
  for (int s = 0; s < STEPS; ++s) {
    const int in = s & 1, outb = in ^ 1;
    k_gru_step<<<256, 256, 0, stream>>>(
        XB[in], 1, decEmb_bf, decWih_bf, decWhh_bf, dec_bih, dec_bhh,
        DF[in], DB[in], DF[outb], DB[outb],
        nullptr, 0, nullptr, nullptr);
    k_classifier<<<BB / 16, 256, 0, stream>>>(
        DB[outb], W1_bf, cls_b1, W2_bf, cls_b2, out, s, XB[outb]);
  }
}
